// MA_Head_24799141167537
// MI455X (gfx1250) — matrix operand order repaired and checked by reading, not compiled, not run on hardware
//
#include <hip/hip_runtime.h>
#include <hip/hip_bf16.h>
#include <math.h>

#define B_ 8
#define T_ 4096
#define E_ 256
#define H_ 256

typedef __attribute__((ext_vector_type(16))) __bf16 v16bf;
typedef __attribute__((ext_vector_type(8)))  float  v8f;

union AFrag { v16bf v; uint4 q[2]; unsigned u[8]; };

#define NEG_BIG (-3.0e38f)

#if defined(__gfx1250__) && __has_builtin(__builtin_amdgcn_global_load_async_to_lds_b128)
#define HAS_ASYNC_LDS 1
#else
#define HAS_ASYNC_LDS 0
#endif

#if HAS_ASYNC_LDS
#define GLOBAL_AS __attribute__((address_space(1)))
#define LDS_AS    __attribute__((address_space(3)))
typedef int v4i_async __attribute__((__vector_size__(4 * sizeof(int))));
__device__ __forceinline__ void async_b128(const void* gsrc, void* ldst) {
    __builtin_amdgcn_global_load_async_to_lds_b128(
        (GLOBAL_AS v4i_async*)(gsrc), (LDS_AS v4i_async*)(ldst), 0, 0);
}
__device__ __forceinline__ void wait_async0() {
#if __has_builtin(__builtin_amdgcn_s_wait_asynccnt)
    __builtin_amdgcn_s_wait_asynccnt(0);
#else
    asm volatile("s_wait_asynccnt 0x0" ::: "memory");
#endif
}
#endif

// Scalar fp32 -> bf16, RNE (cold paths / stores)
__device__ __forceinline__ unsigned short f2bf(float f) {
    union { float f; unsigned u; } x; x.f = f;
    unsigned r = x.u + 0x7FFFu + ((x.u >> 16) & 1u);
    return (unsigned short)(r >> 16);
}

// Packed pair converter: hardware v_cvt_pk_bf16_f32 when available
__device__ __forceinline__ unsigned packbf(float a, float b) {
#if __has_builtin(__builtin_amdgcn_cvt_pk_bf16_f32)
    auto r = __builtin_amdgcn_cvt_pk_bf16_f32(a, b);
    unsigned u;
    __builtin_memcpy(&u, &r, 4);
    return u;
#else
    return (unsigned)f2bf(a) | ((unsigned)f2bf(b) << 16);
#endif
}

// ---------------------------------------------------------------------------
// Kernel 0a: x (fp32) -> xb (bf16), flat. 8 elements / thread.
// ---------------------------------------------------------------------------
__global__ __launch_bounds__(256)
void cvt_x_kernel(const float* __restrict__ x, unsigned short* __restrict__ xb) {
    const long i = ((long)blockIdx.x * 256 + threadIdx.x) * 8;
    float4 a = *(const float4*)(x + i);
    float4 b = *(const float4*)(x + i + 4);
    uint4 o;
    o.x = packbf(a.x, a.y);
    o.y = packbf(a.z, a.w);
    o.z = packbf(b.x, b.y);
    o.w = packbf(b.z, b.w);
    *(uint4*)(xb + i) = o;
}

// ---------------------------------------------------------------------------
// Kernel 0b: W (fp32, [e][h]) -> WT (bf16, [h][e]); Wq pre-scaled by 1/16.
// ---------------------------------------------------------------------------
__global__ __launch_bounds__(256)
void cvt_w_kernel(const float* __restrict__ Wq, const float* __restrict__ Wk,
                  const float* __restrict__ Wv,
                  unsigned short* __restrict__ WTq, unsigned short* __restrict__ WTk,
                  unsigned short* __restrict__ WTv) {
    const int which = blockIdx.y;
    const float* W = (which == 0) ? Wq : ((which == 1) ? Wk : Wv);
    unsigned short* WT = (which == 0) ? WTq : ((which == 1) ? WTk : WTv);
    const float scale = (which == 0) ? 0.0625f : 1.0f;
    const int idx = blockIdx.x * 256 + threadIdx.x;
    const int e = idx >> 8, h = idx & 255;
    WT[h * 256 + e] = f2bf(W[e * 256 + h] * scale);
}

// ---------------------------------------------------------------------------
// Kernel 1: q/k/v projection, all-bf16 operands, every fragment = 2x b128.
// grid = (B*T/16, 3). block = 128 (4 waves); wave computes 16 rows x 64 cols.
// ---------------------------------------------------------------------------
__global__ __launch_bounds__(128)
void qkv_proj_kernel(const unsigned short* __restrict__ xb,
                     const unsigned short* __restrict__ WTq,
                     const unsigned short* __restrict__ WTk,
                     const unsigned short* __restrict__ WTv,
                     unsigned short* __restrict__ Qb,
                     unsigned short* __restrict__ Kb,
                     unsigned short* __restrict__ VTb) {
    const int w    = threadIdx.x >> 5;
    const int lane = threadIdx.x & 31;
    const int li   = lane & 15;
    const int hi   = lane >> 4;
    const int which = blockIdx.y;
    const unsigned short* WT =
        (which == 0) ? WTq : ((which == 1) ? WTk : WTv);
    const long t0 = (long)blockIdx.x * 16;

    v8f acc[4];
    #pragma unroll
    for (int n = 0; n < 4; ++n) {
        v8f z = {0.f, 0.f, 0.f, 0.f, 0.f, 0.f, 0.f, 0.f};
        acc[n] = z;
    }

    #pragma unroll
    for (int kt = 0; kt < 8; ++kt) {
        const int kb = kt * 32;
        AFrag a;  // A: lane li -> row, K chunks {0..7, 16..23} (+8 hi lanes)
        const unsigned short* xrow = xb + (t0 + li) * E_ + kb + hi * 8;
        a.q[0] = *(const uint4*)(xrow);
        a.q[1] = *(const uint4*)(xrow + 16);
        #pragma unroll
        for (int n = 0; n < 4; ++n) {
            const int hb = w * 64 + n * 16 + li;  // B: lane -> column
            AFrag bf;                              // K chunks {0..7, 16..23} (+8 hi lanes)
            const unsigned short* wrow = WT + (long)hb * E_ + kb + hi * 8;
            bf.q[0] = *(const uint4*)(wrow);
            bf.q[1] = *(const uint4*)(wrow + 16);
            acc[n] = __builtin_amdgcn_wmma_f32_16x16x32_bf16(
                false, a.v, false, bf.v, (short)0, acc[n], false, false);
        }
    }

    if (which < 2) {
        unsigned short* dst = (which == 0) ? Qb : Kb;
        #pragma unroll
        for (int n = 0; n < 4; ++n) {
            const int h = w * 64 + n * 16 + li;
            #pragma unroll
            for (int r = 0; r < 8; ++r) {
                const long t = t0 + r + 8 * hi;
                dst[t * H_ + h] = f2bf(acc[n][r]);
            }
        }
    } else {
        #pragma unroll
        for (int n = 0; n < 4; ++n) {
            const int h = w * 64 + n * 16 + li;
            #pragma unroll
            for (int r = 0; r < 8; ++r) {
                const long tg  = t0 + r + 8 * hi;
                const long bb  = tg >> 12;          // / T_
                const long tin = tg & (T_ - 1);
                VTb[(bb * H_ + h) * T_ + tin] = f2bf(acc[n][r]);
            }
        }
    }
}

// ---------------------------------------------------------------------------
// Kernel 2: flash attention. grid = (T/64, B). block = 128 (4 waves).
// Each wave: 16 query rows, full H=256 in registers (16 v8f tiles).
// 32-key blocks DMA'd into LDS via GLOBAL_LOAD_ASYNC_TO_LDS_B128 (ASYNCcnt),
// online softmax, all WMMA fragments = 2x b128.
// ---------------------------------------------------------------------------
__global__ __launch_bounds__(128)
void flash_attn_kernel(const unsigned short* __restrict__ Qb,
                       const unsigned short* __restrict__ Kb,
                       const unsigned short* __restrict__ VTb,
                       float* __restrict__ out) {
    __shared__ unsigned short sK[32 * 256];   // K block, [key][h]
    __shared__ unsigned short sVT[256 * 32];  // V^T block, [h][key]
    __shared__ unsigned short sP[4][16 * 32]; // per-wave P, [row][key]

    const int w    = threadIdx.x >> 5;
    const int lane = threadIdx.x & 31;
    const int li   = lane & 15;
    const int hi   = lane >> 4;
    const int b    = blockIdx.y;
    const int q0   = blockIdx.x * 64;
    const int qw0  = q0 + w * 16;
    const long rowbase = (long)b * T_;

    v8f O[16];
    #pragma unroll
    for (int ct = 0; ct < 16; ++ct) {
        v8f z = {0.f, 0.f, 0.f, 0.f, 0.f, 0.f, 0.f, 0.f};
        O[ct] = z;
    }
    float m[8], l[8];
    #pragma unroll
    for (int r = 0; r < 8; ++r) { m[r] = NEG_BIG; l[r] = 0.f; }

    const int nkb = q0 / 32 + 2;  // keys [0, q0+64)

    for (int kb = 0; kb < nkb; ++kb) {
        const int jb = kb * 32;
        __syncthreads();  // previous iteration done reading LDS

#if HAS_ASYNC_LDS
        {   // K block: 16KB contiguous; 8 async b128 per thread
            const char* srcK = (const char*)(Kb + (rowbase + jb) * H_);
            char* dstK = (char*)sK;
            #pragma unroll
            for (int i = 0; i < 8; ++i) {
                const int off = threadIdx.x * 16 + i * 2048;
                async_b128(srcK + off, dstK + off);
            }
        }
        {   // V^T block: 256 rows x 64B
            #pragma unroll
            for (int i = 0; i < 8; ++i) {
                const int idx = threadIdx.x + i * 128;
                const int h = idx >> 2, part = idx & 3;
                const unsigned short* src =
                    VTb + ((long)b * H_ + h) * T_ + jb + part * 8;
                async_b128(src, (char*)sVT + idx * 16);
            }
        }
        if (kb + 1 < nkb)  // -> global_prefetch_b8
            __builtin_prefetch(Kb + (rowbase + jb + 32) * H_, 0, 1);
        wait_async0();
#else
        {   // fallback: synchronous staging through VGPRs
            const uint4* src = (const uint4*)(Kb + (rowbase + jb) * H_);
            uint4* d = (uint4*)sK;
            #pragma unroll 2
            for (int i = threadIdx.x; i < (32 * 256 * 2) / 16; i += 128)
                d[i] = src[i];
            #pragma unroll 2
            for (int idx = threadIdx.x; idx < 1024; idx += 128) {
                const int h = idx >> 2, part = idx & 3;
                const uint4* s =
                    (const uint4*)(VTb + ((long)b * H_ + h) * T_ + jb) + part;
                ((uint4*)sVT)[idx] = *s;
            }
        }
        if (kb + 1 < nkb)
            __builtin_prefetch(Kb + (rowbase + jb + 32) * H_, 0, 1);
#endif
        __syncthreads();

        // --- scores: two 16x16 tiles, K-loop over H=256 ---
        v8f s0 = {0.f, 0.f, 0.f, 0.f, 0.f, 0.f, 0.f, 0.f};
        v8f s1 = s0;
        #pragma unroll
        for (int kt = 0; kt < 8; ++kt) {
            AFrag qf, kf0, kf1;
            const unsigned short* qrow =
                Qb + (rowbase + qw0 + li) * H_ + kt * 32 + hi * 8;
            qf.q[0] = *(const uint4*)(qrow);
            qf.q[1] = *(const uint4*)(qrow + 16);
            const unsigned short* krow0 = sK + li * 256 + kt * 32 + hi * 8;
            const unsigned short* krow1 = krow0 + 16 * 256;
            kf0.q[0] = *(const uint4*)(krow0);
            kf0.q[1] = *(const uint4*)(krow0 + 16);
            kf1.q[0] = *(const uint4*)(krow1);
            kf1.q[1] = *(const uint4*)(krow1 + 16);
            s0 = __builtin_amdgcn_wmma_f32_16x16x32_bf16(
                false, qf.v, false, kf0.v, (short)0, s0, false, false);
            s1 = __builtin_amdgcn_wmma_f32_16x16x32_bf16(
                false, qf.v, false, kf1.v, (short)0, s1, false, false);
        }

        // --- causal mask + online softmax ---
        float alpha[8], p0s[8], p1s[8];
        #pragma unroll
        for (int r = 0; r < 8; ++r) {
            const int qg = qw0 + r + 8 * hi;
            const float v0 = ((jb + li)      <= qg) ? s0[r] : NEG_BIG;
            const float v1 = ((jb + 16 + li) <= qg) ? s1[r] : NEG_BIG;
            float tm = fmaxf(v0, v1);
            #pragma unroll
            for (int off = 1; off < 16; off <<= 1)
                tm = fmaxf(tm, __shfl_xor(tm, off, 32));
            const float mn = fmaxf(m[r], tm);
            const float al = __expf(m[r] - mn);
            const float p0 = __expf(v0 - mn);
            const float p1 = __expf(v1 - mn);
            float ts = p0 + p1;
            #pragma unroll
            for (int off = 1; off < 16; off <<= 1)
                ts += __shfl_xor(ts, off, 32);
            l[r] = l[r] * al + ts;
            m[r] = mn;
            alpha[r] = al;
            p0s[r] = p0;
            p1s[r] = p1;
        }

        // --- D-layout -> A-layout via LDS ---
        #pragma unroll
        for (int r = 0; r < 8; ++r) {
            const int row = r + 8 * hi;
            sP[w][row * 32 + li]      = f2bf(p0s[r]);
            sP[w][row * 32 + 16 + li] = f2bf(p1s[r]);
        }
        __syncthreads();

        // --- rescale O, then P @ V^T (K=32) ---
        #pragma unroll
        for (int ct = 0; ct < 16; ++ct)
            #pragma unroll
            for (int r = 0; r < 8; ++r)
                O[ct][r] *= alpha[r];

        AFrag pf;
        const unsigned short* prow = &sP[w][li * 32 + hi * 8];
        pf.q[0] = *(const uint4*)(prow);
        pf.q[1] = *(const uint4*)(prow + 16);

        #pragma unroll
        for (int ct = 0; ct < 16; ++ct) {
            AFrag vf;
            const unsigned short* vrow = sVT + (ct * 16 + li) * 32 + hi * 8;
            vf.q[0] = *(const uint4*)(vrow);
            vf.q[1] = *(const uint4*)(vrow + 16);
            O[ct] = __builtin_amdgcn_wmma_f32_16x16x32_bf16(
                false, pf.v, false, vf.v, (short)0, O[ct], false, false);
        }
    }

    // --- epilogue ---
    float rinv[8];
    #pragma unroll
    for (int r = 0; r < 8; ++r) rinv[r] = 1.0f / l[r];
    #pragma unroll
    for (int ct = 0; ct < 16; ++ct) {
        const int h = ct * 16 + li;
        #pragma unroll
        for (int r = 0; r < 8; ++r) {
            const long t = rowbase + qw0 + r + 8 * hi;
            out[t * H_ + h] = O[ct][r] * rinv[r];
        }
    }
}

// ---------------------------------------------------------------------------
extern "C" void kernel_launch(void* const* d_in, const int* in_sizes, int n_in,
                              void* d_out, int out_size, void* d_ws, size_t ws_size,
                              hipStream_t stream) {
    // setup_inputs order: x, Wk, Wq, Wv (all fp32)
    const float* x  = (const float*)d_in[0];
    const float* Wk = (const float*)d_in[1];
    const float* Wq = (const float*)d_in[2];
    const float* Wv = (const float*)d_in[3];

    // Workspace (bf16 elements):
    // Q[B*T][H], K[B*T][H], VT[B][H][T], xb[B*T][E], WTq/WTk/WTv[H][E]
    unsigned short* Qb  = (unsigned short*)d_ws;
    unsigned short* Kb  = Qb  + (size_t)B_ * T_ * H_;
    unsigned short* VTb = Kb  + (size_t)B_ * T_ * H_;
    unsigned short* xbb = VTb + (size_t)B_ * T_ * H_;
    unsigned short* WTq = xbb + (size_t)B_ * T_ * E_;
    unsigned short* WTk = WTq + (size_t)E_ * H_;
    unsigned short* WTv = WTk + (size_t)E_ * H_;

    const long nx = (long)B_ * T_ * E_;
    cvt_x_kernel<<<dim3(nx / (256 * 8)), 256, 0, stream>>>(x, xbb);
    cvt_w_kernel<<<dim3(256, 3), 256, 0, stream>>>(Wq, Wk, Wv, WTq, WTk, WTv);

    dim3 g1((B_ * T_) / 16, 3, 1);
    qkv_proj_kernel<<<g1, 128, 0, stream>>>(xbb, WTq, WTk, WTv, Qb, Kb, VTb);

    dim3 g2(T_ / 64, B_, 1);
    flash_attn_kernel<<<g2, 128, 0, stream>>>(Qb, Kb, VTb, (float*)d_out);
}
